// EdgeGuidedCrossAttention_25975962206500
// MI455X (gfx1250) — matrix operand order repaired and checked by reading, not compiled, not run on hardware
//
#include <hip/hip_runtime.h>
#include <math.h>

#define H 128

typedef __attribute__((ext_vector_type(16))) _Float16 v16h;
typedef __attribute__((ext_vector_type(8)))  float    v8f;
typedef __attribute__((ext_vector_type(4)))  _Float16 v4h;
typedef __attribute__((ext_vector_type(4)))  float    v4f;

static __device__ __forceinline__ float wave_sum(float v) {
#pragma unroll
  for (int off = 16; off > 0; off >>= 1) v += __shfl_xor(v, off, 32);
  return v;
}
static __device__ __forceinline__ float wave_max(float v) {
#pragma unroll
  for (int off = 16; off > 0; off >>= 1) v = fmaxf(v, __shfl_xor(v, off, 32));
  return v;
}
// order-preserving float <-> uint for atomicMax
static __device__ __forceinline__ unsigned flipF(float f) {
  unsigned u = __float_as_uint(f);
  return u ^ ((unsigned)((int)u >> 31) | 0x80000000u);
}
static __device__ __forceinline__ float unflipF(unsigned u) {
  unsigned mask = (u & 0x80000000u) ? 0x80000000u : 0xFFFFFFFFu;
  return __uint_as_float(u ^ mask);
}

// ---------------- f32 -> f16 weight preconversion (once per call) ----------------
__global__ __launch_bounds__(256) void cvt_f16_kernel(const float* __restrict__ src,
                                                      _Float16* __restrict__ dst, int n) {
  int stride = gridDim.x * blockDim.x;
  for (int i = blockIdx.x * blockDim.x + threadIdx.x; i < n; i += stride)
    dst[i] = (_Float16)src[i];
}

// ---------------- edge bias: bias[e] = ea[e,:] . w ----------------
__global__ __launch_bounds__(256) void bias_kernel(const float* __restrict__ ea,
                                                   const float* __restrict__ w,
                                                   float* __restrict__ bias, int E, int CE) {
  int stride = gridDim.x * blockDim.x;
  for (int e = blockIdx.x * blockDim.x + threadIdx.x; e < E; e += stride) {
    const float* row = ea + (size_t)e * CE;
    float s = 0.f;
    for (int i = 0; i < CE; ++i) s += row[i] * w[i];
    bias[e] = s;
  }
}

// ---------------- Y = X @ W^T + b via v_wmma_f32_16x16x32_f16 ----------------
// W16 is the pre-converted f16 copy of W (row-major [H,H], row = output col).
// One wave computes a 16-row x 128-col tile. A is fully loaded to registers
// before compute, so Y may alias X (used for the in-place output projection).
template <typename OutT>
__global__ __launch_bounds__(256) void gemm_xwT(const float* __restrict__ X,
                                                const _Float16* __restrict__ W16,
                                                const float* __restrict__ bvec,
                                                OutT* __restrict__ Y, int N) {
  int lane = threadIdx.x & 31;
  int wave = threadIdx.x >> 5;
  int row0 = blockIdx.x * 128 + wave * 16;
  if (row0 >= N) return;                    // wave-uniform exit, EXEC stays all-1 for WMMA
  int half = lane >> 4;                     // 0: lanes 0-15, 1: lanes 16-31
  int m    = lane & 15;

  // A fragments: 16x32 f16, contiguous K order (matches B)
  //   lanes 0-15 : halves 0..15 = K k0+0..15
  //   lanes 16-31: halves 0..15 = K k0+16..31
  int arow = row0 + m; if (arow >= N) arow = N - 1;
  const float* xrow = X + (size_t)arow * H;
  v16h a[4];
#pragma unroll
  for (int kb = 0; kb < 4; ++kb) {
    const float* p0 = xrow + kb * 32 + half * 16;
    const float* p1 = xrow + kb * 32 + half * 16 + 8;
#pragma unroll
    for (int j = 0; j < 8; ++j) {
      a[kb][j]     = (_Float16)p0[j];
      a[kb][j + 8] = (_Float16)p1[j];
    }
  }

#pragma unroll
  for (int nt = 0; nt < 8; ++nt) {
    int col = nt * 16 + m;
    float bn = bvec[col];
    v8f c = {bn, bn, bn, bn, bn, bn, bn, bn};
#pragma unroll
    for (int kb = 0; kb < 4; ++kb) {
      // B fragment 32x16: lane&15 = column N (= W row), lanes 0-15 hold K 0..15,
      // lanes 16-31 hold K 16..31 (16 contiguous halves per lane, 32B aligned)
      v16h bf = *(const v16h*)(W16 + (size_t)col * H + kb * 32 + half * 16);
      c = __builtin_amdgcn_wmma_f32_16x16x32_f16(false, a[kb], false, bf,
                                                 (short)0, c, false, false);
    }
#pragma unroll
    for (int r = 0; r < 8; ++r) {
      int row = row0 + r + 8 * half;        // C/D layout: VGPR r -> M = r + 8*half
      if (row < N) Y[(size_t)row * H + col] = (OutT)c[r];
    }
  }
}

// ------------- per-edge scores + global max (wave per edge) -------------
__global__ __launch_bounds__(256) void edge_score_kernel(
    const _Float16* __restrict__ Qb, const _Float16* __restrict__ Kb,
    const int* __restrict__ qidx, const int* __restrict__ kidx,
    const float* __restrict__ bias, float* __restrict__ scores,
    unsigned* __restrict__ maxslot, int E, float scale) {
  __shared__ float smax[8];
  int lane = threadIdx.x & 31;
  int wid  = (blockIdx.x * blockDim.x + threadIdx.x) >> 5;
  int nw   = (gridDim.x * blockDim.x) >> 5;
  float lmax = -3.4e38f;
  for (int e = wid; e < E; e += nw) {
    v4h qv = *((const v4h*)(Qb + (size_t)qidx[e] * H) + lane);
    v4h kv = *((const v4h*)(Kb + (size_t)kidx[e] * H) + lane);
    float d = (float)qv[0] * (float)kv[0] + (float)qv[1] * (float)kv[1] +
              (float)qv[2] * (float)kv[2] + (float)qv[3] * (float)kv[3];
    d = wave_sum(d);
    float s = d * scale + bias[e];
    if (lane == 0) scores[e] = s;
    lmax = fmaxf(lmax, s);
  }
  lmax = wave_max(lmax);
  if (lane == 0) smax[threadIdx.x >> 5] = lmax;
  __syncthreads();
  if (threadIdx.x == 0) {
    float mx = smax[0];
    for (int i = 1; i < (int)(blockDim.x >> 5); ++i) mx = fmaxf(mx, smax[i]);
    atomicMax(maxslot, flipF(mx));
  }
}

// ------------- exp + segment denominator -------------
__global__ __launch_bounds__(256) void exp_denom_kernel(
    float* __restrict__ scores, const int* __restrict__ tgt,
    float* __restrict__ denom, const unsigned* __restrict__ maxslot, int E) {
  float gmax = unflipF(*maxslot);
  int stride = gridDim.x * blockDim.x;
  for (int e = blockIdx.x * blockDim.x + threadIdx.x; e < E; e += stride) {
    float w = __expf(scores[e] - gmax);
    scores[e] = w;                           // in-place: scores buffer becomes exp_s
    unsafeAtomicAdd(&denom[tgt[e]], w);
  }
}

// ------------- weighted scatter-add of V rows (wave per edge) -------------
__global__ __launch_bounds__(256) void scatter_kernel(
    const float* __restrict__ exps, const float* __restrict__ denom,
    const _Float16* __restrict__ Vb, const int* __restrict__ vidx,
    const int* __restrict__ tgt, float* __restrict__ upd, int E) {
  int lane = threadIdx.x & 31;
  int wid  = (blockIdx.x * blockDim.x + threadIdx.x) >> 5;
  int nw   = (gridDim.x * blockDim.x) >> 5;
  for (int e = wid; e < E; e += nw) {
    int t = tgt[e];
    float coef = exps[e] / (denom[t] + 1e-8f);
    v4h vv = *((const v4h*)(Vb + (size_t)vidx[e] * H) + lane);
    float* dst = upd + (size_t)t * H + lane * 4;
    unsafeAtomicAdd(dst + 0, coef * (float)vv[0]);
    unsafeAtomicAdd(dst + 1, coef * (float)vv[1]);
    unsafeAtomicAdd(dst + 2, coef * (float)vv[2]);
    unsafeAtomicAdd(dst + 3, coef * (float)vv[3]);
  }
}

// ------------- residual + LayerNorm (wave per row) -------------
__global__ __launch_bounds__(256) void ln_kernel(
    const float* __restrict__ x, const float* __restrict__ y,
    const float* __restrict__ g, const float* __restrict__ be,
    float* __restrict__ out, int N) {
  int lane = threadIdx.x & 31;
  int wid  = (blockIdx.x * blockDim.x + threadIdx.x) >> 5;
  int nw   = (gridDim.x * blockDim.x) >> 5;
  for (int row = wid; row < N; row += nw) {
    v4f xv = *((const v4f*)(x + (size_t)row * H) + lane);
    v4f yv = *((const v4f*)(y + (size_t)row * H) + lane);
    v4f v  = xv + yv;
    float s = v[0] + v[1] + v[2] + v[3];
    float q = v[0] * v[0] + v[1] * v[1] + v[2] * v[2] + v[3] * v[3];
    s = wave_sum(s);
    q = wave_sum(q);
    float mu  = s * (1.0f / H);
    float var = q * (1.0f / H) - mu * mu;
    float rs  = rsqrtf(var + 1e-5f);
    v4f gv = *((const v4f*)g + lane);
    v4f bv = *((const v4f*)be + lane);
    v4f o;
    o[0] = (v[0] - mu) * rs * gv[0] + bv[0];
    o[1] = (v[1] - mu) * rs * gv[1] + bv[1];
    o[2] = (v[2] - mu) * rs * gv[2] + bv[2];
    o[3] = (v[3] - mu) * rs * gv[3] + bv[3];
    *((v4f*)(out + (size_t)row * H) + lane) = o;
  }
}

extern "C" void kernel_launch(void* const* d_in, const int* in_sizes, int n_in,
                              void* d_out, int out_size, void* d_ws, size_t ws_size,
                              hipStream_t stream) {
  const float* pro_x = (const float*)d_in[0];
  const float* lig_x = (const float*)d_in[1];
  const int*   ei    = (const int*)d_in[2];
  const float* ea    = (const float*)d_in[3];
  const float *Wq_pro = (const float*)d_in[4],  *bq_pro = (const float*)d_in[5];
  const float *Wk_lig = (const float*)d_in[6],  *bk_lig = (const float*)d_in[7];
  const float *Wv_lig = (const float*)d_in[8],  *bv_lig = (const float*)d_in[9];
  const float *Wq_lig = (const float*)d_in[10], *bq_lig = (const float*)d_in[11];
  const float *Wk_pro = (const float*)d_in[12], *bk_pro = (const float*)d_in[13];
  const float *Wv_pro = (const float*)d_in[14], *bv_pro = (const float*)d_in[15];
  const float *We_bias = (const float*)d_in[16];
  const float *Wo_pro = (const float*)d_in[17], *bo_pro = (const float*)d_in[18];
  const float *Wo_lig = (const float*)d_in[19], *bo_lig = (const float*)d_in[20];
  const float *g_pro = (const float*)d_in[21], *be_pro = (const float*)d_in[22];
  const float *g_lig = (const float*)d_in[23], *be_lig = (const float*)d_in[24];

  int NP = in_sizes[0] / H;
  int NL = in_sizes[1] / H;
  int E  = in_sizes[2] / 2;
  int CE = in_sizes[3] / E;
  const int* pi = ei;
  const int* li = ei + E;

  // ---- workspace carve-up (256B aligned blocks) ----
  char* p = (char*)d_ws;
  auto carve = [&](size_t bytes) -> char* {
    char* r = p;
    p += (bytes + 255) & ~(size_t)255;
    return r;
  };
  _Float16* Qp = (_Float16*)carve((size_t)NP * H * 2);
  _Float16* Kp = (_Float16*)carve((size_t)NP * H * 2);
  _Float16* Vp = (_Float16*)carve((size_t)NP * H * 2);
  _Float16* Ql = (_Float16*)carve((size_t)NL * H * 2);
  _Float16* Kl = (_Float16*)carve((size_t)NL * H * 2);
  _Float16* Vl = (_Float16*)carve((size_t)NL * H * 2);
  float*    bias = (float*)carve((size_t)E * 4);
  float*    exps = (float*)carve((size_t)E * 4);      // raw scores, then exp_s
  _Float16* W16  = (_Float16*)carve((size_t)8 * H * H * 2);  // 8 pre-converted weights
  char*     zbeg = p;                                  // zero-init region starts here
  float*    denom_p = (float*)carve((size_t)NP * 4);
  float*    denom_l = (float*)carve((size_t)NL * 4);
  unsigned* maxbuf  = (unsigned*)carve(256);
  float*    upd_p   = (float*)carve((size_t)NP * H * 4);
  float*    upd_l   = (float*)carve((size_t)NL * H * 4);
  hipMemsetAsync(zbeg, 0, (size_t)(p - zbeg), stream);

  _Float16* Wq_pro16 = W16 + 0 * H * H;
  _Float16* Wk_pro16 = W16 + 1 * H * H;
  _Float16* Wv_pro16 = W16 + 2 * H * H;
  _Float16* Wq_lig16 = W16 + 3 * H * H;
  _Float16* Wk_lig16 = W16 + 4 * H * H;
  _Float16* Wv_lig16 = W16 + 5 * H * H;
  _Float16* Wo_pro16 = W16 + 6 * H * H;
  _Float16* Wo_lig16 = W16 + 7 * H * H;

  float* out_pro = (float*)d_out;
  float* out_lig = (float*)d_out + (size_t)NP * H;
  float scale = 1.0f / sqrtf((float)H);

  dim3 blk(256);
  int gp = (NP + 127) / 128, gl = (NL + 127) / 128;
  int nW = H * H;

  // pre-convert the 8 weight matrices to f16 (32KB each in f16, all waves reuse from L2/WGP$)
  cvt_f16_kernel<<<16, blk, 0, stream>>>(Wq_pro, Wq_pro16, nW);
  cvt_f16_kernel<<<16, blk, 0, stream>>>(Wk_pro, Wk_pro16, nW);
  cvt_f16_kernel<<<16, blk, 0, stream>>>(Wv_pro, Wv_pro16, nW);
  cvt_f16_kernel<<<16, blk, 0, stream>>>(Wq_lig, Wq_lig16, nW);
  cvt_f16_kernel<<<16, blk, 0, stream>>>(Wk_lig, Wk_lig16, nW);
  cvt_f16_kernel<<<16, blk, 0, stream>>>(Wv_lig, Wv_lig16, nW);
  cvt_f16_kernel<<<16, blk, 0, stream>>>(Wo_pro, Wo_pro16, nW);
  cvt_f16_kernel<<<16, blk, 0, stream>>>(Wo_lig, Wo_lig16, nW);

  bias_kernel<<<512, blk, 0, stream>>>(ea, We_bias, bias, E, CE);

  gemm_xwT<_Float16><<<gp, blk, 0, stream>>>(pro_x, Wq_pro16, bq_pro, Qp, NP);
  gemm_xwT<_Float16><<<gp, blk, 0, stream>>>(pro_x, Wk_pro16, bk_pro, Kp, NP);
  gemm_xwT<_Float16><<<gp, blk, 0, stream>>>(pro_x, Wv_pro16, bv_pro, Vp, NP);
  gemm_xwT<_Float16><<<gl, blk, 0, stream>>>(lig_x, Wq_lig16, bq_lig, Ql, NL);
  gemm_xwT<_Float16><<<gl, blk, 0, stream>>>(lig_x, Wk_lig16, bk_lig, Kl, NL);
  gemm_xwT<_Float16><<<gl, blk, 0, stream>>>(lig_x, Wv_lig16, bv_lig, Vl, NL);

  // direction 0: protein <- ligand  (Q = Qp[pi], K = Kl[li], V = Vl[li], tgt = pi)
  edge_score_kernel<<<1024, blk, 0, stream>>>(Qp, Kl, pi, li, bias, exps, maxbuf + 0, E, scale);
  exp_denom_kernel<<<1024, blk, 0, stream>>>(exps, pi, denom_p, maxbuf + 0, E);
  scatter_kernel<<<2048, blk, 0, stream>>>(exps, denom_p, Vl, li, pi, upd_p, E);

  // direction 1: ligand <- protein  (Q = Ql[li], K = Kp[pi], V = Vp[pi], tgt = li)
  edge_score_kernel<<<1024, blk, 0, stream>>>(Ql, Kp, li, pi, bias, exps, maxbuf + 1, E, scale);
  exp_denom_kernel<<<1024, blk, 0, stream>>>(exps, li, denom_l, maxbuf + 1, E);
  scatter_kernel<<<2048, blk, 0, stream>>>(exps, denom_l, Vp, pi, li, upd_l, E);

  // output projections (in-place: A rows fully register-resident before store)
  gemm_xwT<float><<<gp, blk, 0, stream>>>(upd_p, Wo_pro16, bo_pro, upd_p, NP);
  gemm_xwT<float><<<gl, blk, 0, stream>>>(upd_l, Wo_lig16, bo_lig, upd_l, NL);

  ln_kernel<<<1024, blk, 0, stream>>>(pro_x, upd_p, g_pro, be_pro, out_pro, NP);
  ln_kernel<<<1024, blk, 0, stream>>>(lig_x, upd_l, g_lig, be_lig, out_lig, NL);
}
